// PositionEncoding_7473243095307
// MI455X (gfx1250) — matrix operand order repaired and checked by reading, not compiled, not run on hardware
//
#include <hip/hip_runtime.h>

typedef __attribute__((ext_vector_type(16))) _Float16 v16h;
typedef __attribute__((ext_vector_type(8)))  float    v8f;

#define BB 4
#define CC 3
#define NN 32768
#define KKN 16                      // knn K
#define OO 64
#define PTS_PER_B (NN * KKN)        // 524288 points per batch
#define TOTAL_TILES (BB * NN)       // one tile = one n (its 16 neighbors) = 131072
#define WPB 8                       // waves per block
#define TPW 8                       // tiles per wave

__global__ __launch_bounds__(256) void pe_mlp_kernel(
    const float* __restrict__ coords,   // (B, C, N) fp32
    const int*   __restrict__ knn32,    // (B, N, K) int64 viewed as int pairs (LE low dword)
    const float* __restrict__ W1,       // (64, 3)
    const float* __restrict__ b1,       // (64)
    const float* __restrict__ W2,       // (64, 64)
    const float* __restrict__ b2,       // (64)
    float* __restrict__ out)            // (B, 64, N, K) fp32
{
    __shared__ float lw1[OO * 4];       // row o: {w0, w1, w2, b1}
    __shared__ float lw2[OO * OO];
    __shared__ float lb2[OO];

    const int tid = threadIdx.x;
    if (tid < OO) {
        lw1[tid * 4 + 0] = W1[tid * 3 + 0];
        lw1[tid * 4 + 1] = W1[tid * 3 + 1];
        lw1[tid * 4 + 2] = W1[tid * 3 + 2];
        lw1[tid * 4 + 3] = b1[tid];
        lb2[tid]         = b2[tid];
    }
    for (int i = tid; i < OO * OO; i += 256) lw2[i] = W2[i];
    __syncthreads();

    const int lane = tid & 31;
    const int nl   = lane & 15;   // point (column) index within tile
    const int g    = lane >> 4;   // half-wave group

    // ---- Hoisted W2 A-fragments (f16, 16x32 per m-tile/k-step), ISA 7.12.2 layout:
    // VGPR v<4: K = 2v + r + 8g ; v>=4: K = 16 + 2(v-4) + r + 8g  (r = half)
    v16h A[4][2];
    #pragma unroll
    for (int m = 0; m < 4; ++m)
        #pragma unroll
        for (int s = 0; s < 2; ++s)
            #pragma unroll
            for (int e = 0; e < 16; ++e) {
                const int k = ((e >> 3) << 4) + (((e >> 1) & 3) << 1) + (e & 1) + (g << 3);
                A[m][s][e] = (_Float16)lw2[(16 * m + nl) * OO + 32 * s + k];
            }

    // ---- b2 folded into the accumulator init: C/D layout lane holds (M = v + 8g, N = nl)
    v8f bias[4];
    #pragma unroll
    for (int m = 0; m < 4; ++m)
        #pragma unroll
        for (int v = 0; v < 8; ++v)
            bias[m][v] = lb2[16 * m + 8 * g + v];

    const int wave = blockIdx.x * WPB + (tid >> 5);

    for (int i = 0; i < TPW; ++i) {
        const int t = wave * TPW + i;          // wave-uniform
        if (t >= TOTAL_TILES) break;           // wave-uniform: EXEC stays all-ones
        const int b   = t >> 15;               // t / NN
        const int npt = t & (NN - 1);

        const float* cb = coords + (size_t)b * (CC * NN);
        // int64 knn index: low dword (values < 32768)
        const int idx = knn32[2 * (((size_t)b * NN + npt) * KKN + nl)];

        const float x0 = cb[npt]          - cb[idx];
        const float x1 = cb[NN + npt]     - cb[NN + idx];
        const float x2 = cb[2*NN + npt]   - cb[2*NN + idx];

        // ---- Layer 1 (fp32 FMA) computed directly in B-fragment order:
        // lane (nl, g) element e holds h[o = 32s + 8g + (e&7) + 16(e>>3)] for point nl.
        v16h B0, B1;
        #pragma unroll
        for (int e = 0; e < 16; ++e) {
            const float4 wa = *(const float4*)&lw1[(8 * g + (e & 7) + 16 * (e >> 3)) * 4];        // s = 0
            float h0 = fmaf(wa.x, x0, fmaf(wa.y, x1, fmaf(wa.z, x2, wa.w)));
            B0[e] = (_Float16)fmaxf(h0, 0.0f);
            const float4 wb = *(const float4*)&lw1[(32 + 8 * g + (e & 7) + 16 * (e >> 3)) * 4];   // s = 1
            float h1 = fmaf(wb.x, x0, fmaf(wb.y, x1, fmaf(wb.z, x2, wb.w)));
            B1[e] = (_Float16)fmaxf(h1, 0.0f);
        }

        // ---- Layer 2: 4 m-tiles x 2 k-steps of v_wmma_f32_16x16x32_f16
        float* ob = out + (size_t)b * ((size_t)OO * PTS_PER_B) + (size_t)npt * KKN + nl;
        #pragma unroll
        for (int m = 0; m < 4; ++m) {
            v8f acc = bias[m];
            acc = __builtin_amdgcn_wmma_f32_16x16x32_f16(
                false, A[m][0], false, B0, (short)0, acc, false, false);
            acc = __builtin_amdgcn_wmma_f32_16x16x32_f16(
                false, A[m][1], false, B1, (short)0, acc, false, false);
            #pragma unroll
            for (int v = 0; v < 8; ++v)
                __builtin_nontemporal_store(
                    acc[v], ob + (size_t)(16 * m + 8 * g + v) * PTS_PER_B);
        }
    }
}

extern "C" void kernel_launch(void* const* d_in, const int* in_sizes, int n_in,
                              void* d_out, int out_size, void* d_ws, size_t ws_size,
                              hipStream_t stream) {
    (void)in_sizes; (void)n_in; (void)out_size; (void)d_ws; (void)ws_size;
    const float* coords = (const float*)d_in[0];
    const int*   knn32  = (const int*)d_in[1];   // int64 data, read low dwords
    const float* W1     = (const float*)d_in[2];
    const float* b1     = (const float*)d_in[3];
    const float* W2     = (const float*)d_in[4];
    const float* b2     = (const float*)d_in[5];
    float* out = (float*)d_out;

    const int blocks = TOTAL_TILES / (WPB * TPW);   // 2048
    pe_mlp_kernel<<<blocks, 256, 0, stream>>>(coords, knn32, W1, b1, W2, b2, out);
}
